// Multi_Line_Model_on_going_2_78297253806566
// MI455X (gfx1250) — compile-verified, matrix operand order checked by reading, not run on hardware
//
#include <hip/hip_runtime.h>
#include <hip/hip_bf16.h>

// ---------------------------------------------------------------------------
// Problem constants (from reference)
// ---------------------------------------------------------------------------
#define BB   128
#define CC   512
#define HWN  256          // H*W
#define HID  512
#define NL   16           // n_lines
#define MAXPT 40
#define OUTD 42           // MAX_PT_DIM + 2
#define INL  554          // C + OUT
#define KCMB 1088         // padded K for [x(554) | h(512) | pad(22)]
#define N4H  2048         // 4*HID
#define OUTP 48           // padded OUT (3 N-tiles)

typedef __attribute__((ext_vector_type(16))) _Float16 v16h;
typedef __attribute__((ext_vector_type(8)))  _Float16 v8h;
typedef __attribute__((ext_vector_type(8)))  float    v8f;

// ---------------------------------------------------------------------------
// Exact JAX threefry2x32 for key(7) -> uniform (16,128); index n = t*128 + b.
// counts = iota(2048) split in halves: n<1024 -> (n, n+1024) word0,
// n>=1024 -> (n-1024, n) word1.
// ---------------------------------------------------------------------------
__device__ __forceinline__ unsigned rotl32(unsigned x, int r) {
    return (x << r) | (x >> (32 - r));
}
__device__ __forceinline__ float jax_uniform_tb(int t, int b) {
    unsigned n = (unsigned)(t * BB + b);
    unsigned x0, x1;
    int word;
    if (n < 1024u) { x0 = n; x1 = n + 1024u; word = 0; }
    else           { x0 = n - 1024u; x1 = n; word = 1; }
    unsigned ks[3] = {0u, 7u, 0x1BD11BDAu ^ 0u ^ 7u};
    unsigned x[2] = {x0 + ks[0], x1 + ks[1]};
    const int rotA[4] = {13, 15, 26, 6};
    const int rotB[4] = {17, 29, 16, 24};
    #pragma unroll
    for (int blk = 0; blk < 5; ++blk) {
        const int* R = (blk & 1) ? rotB : rotA;
        #pragma unroll
        for (int j = 0; j < 4; ++j) {
            x[0] += x[1];
            x[1] = rotl32(x[1], R[j]);
            x[1] ^= x[0];
        }
        x[0] += ks[(blk + 1) % 3];
        x[1] += ks[(blk + 2) % 3] + (unsigned)(blk + 1);
    }
    unsigned bits = x[word];
    unsigned fb = (bits >> 9) | 0x3F800000u;
    return __uint_as_float(fb) - 1.0f;
}

// ---------------------------------------------------------------------------
// Wave (32-lane) reductions
// ---------------------------------------------------------------------------
__device__ __forceinline__ float wave_sum(float v) {
    #pragma unroll
    for (int off = 16; off > 0; off >>= 1) v += __shfl_xor(v, off, 32);
    return v;
}
__device__ __forceinline__ float wave_max(float v) {
    #pragma unroll
    for (int off = 16; off > 0; off >>= 1) v = fmaxf(v, __shfl_xor(v, off, 32));
    return v;
}

// ---------------------------------------------------------------------------
// WMMA multi-tile core: one 16xK A panel row-block (row-major f16, lda) times
// NT adjacent 16-column blocks of W (N x K row-major f16, ldb), f32 accumulate.
// A fragment is loaded once per K-step and reused across NT wmma ops.
// Per ISA 7.12.2 16-bit layouts: lane l holds row m0+(l&15); half-chunks at
// K offsets {0,16} (lanes 0-15) / {8,24} (lanes 16-31). B mirrors with N.
// ---------------------------------------------------------------------------
template <int NT>
__device__ __forceinline__ void wmma_panel(const _Float16* __restrict__ A,
                                           const _Float16* __restrict__ W,
                                           int m0, int n0, int K, int lda, int ldb,
                                           v8f (&acc)[NT]) {
    const int lane  = threadIdx.x & 31;
    const int koff0 = (lane < 16) ? 0 : 8;
    const int koff1 = (lane < 16) ? 16 : 24;
    const _Float16* arp = A + (size_t)(m0 + (lane & 15)) * lda;
    const _Float16* brp[NT];
    #pragma unroll
    for (int j = 0; j < NT; ++j)
        brp[j] = W + (size_t)(n0 + j * 16 + (lane & 15)) * ldb;
    #pragma unroll
    for (int j = 0; j < NT; ++j) acc[j] = (v8f){};
    for (int k0 = 0; k0 < K; k0 += 32) {
        union { v16h v; v8h h[2]; } a;
        a.h[0] = *(const v8h*)(arp + k0 + koff0);
        a.h[1] = *(const v8h*)(arp + k0 + koff1);
        #pragma unroll
        for (int j = 0; j < NT; ++j) {
            union { v16h v; v8h h[2]; } b;
            b.h[0] = *(const v8h*)(brp[j] + k0 + koff0);
            b.h[1] = *(const v8h*)(brp[j] + k0 + koff1);
            acc[j] = __builtin_amdgcn_wmma_f32_16x16x32_f16(
                false, a.v, false, b.v, (short)0, acc[j], false, false);
        }
    }
}

// ---------------------------------------------------------------------------
// Prep kernels: pack weights to f16
// ---------------------------------------------------------------------------
__global__ void conv_wc_k(const float* __restrict__ wih,
                          const float* __restrict__ whh,
                          _Float16* __restrict__ wc) {
    int idx = blockIdx.x * 256 + threadIdx.x;
    if (idx >= N4H * KCMB) return;
    int n = idx / KCMB;
    int k = idx - n * KCMB;
    float v;
    if (k < INL)              v = wih[n * INL + k];
    else if (k < INL + HID)   v = whh[n * HID + (k - INL)];
    else                      v = 0.0f;
    wc[idx] = (_Float16)v;
}

__global__ void conv_f16_k(const float* __restrict__ src,
                           _Float16* __restrict__ dst, int n) {
    int idx = blockIdx.x * 256 + threadIdx.x;
    if (idx < n) dst[idx] = (_Float16)src[idx];
}

__global__ void conv_l2_k(const float* __restrict__ l2w,
                          _Float16* __restrict__ dst) {
    int idx = blockIdx.x * 256 + threadIdx.x;
    if (idx >= OUTP * HID) return;
    int n = idx / HID;
    int k = idx - n * HID;
    dst[idx] = (_Float16)((n < OUTD) ? l2w[n * HID + k] : 0.0f);
}

__global__ void zero_k(float* __restrict__ p, int n) {
    int idx = blockIdx.x * 256 + threadIdx.x;
    if (idx < n) p[idx] = 0.0f;
}

// s_img[b][n] = sum_c feat[b][c][n] * w_img[c] + att_b
__global__ void simg_k(const float* __restrict__ feat,
                       const float* __restrict__ w_img,
                       const float* __restrict__ att_b,
                       float* __restrict__ simg) {
    int b = blockIdx.x, n = threadIdx.x;
    const float* fb = feat + (size_t)b * CC * HWN;
    float acc = *att_b;
    for (int c = 0; c < CC; ++c) acc += fb[c * HWN + n] * w_img[c];
    simg[b * HWN + n] = acc;
}

// ---------------------------------------------------------------------------
// Fused per-step attention: hdot -> softmax -> pooled, plus scheduled-sampling
// input and h copy into the combined A operand xh (B x KCMB, f16).
// One block per batch element, 256 threads (8 waves).
// ---------------------------------------------------------------------------
__global__ void attn_k(const float* __restrict__ feat,
                       const float* __restrict__ simg,
                       const float* __restrict__ w_hid,
                       const float* __restrict__ h,
                       const float* __restrict__ outs,   // (B,16,OUTP) f32
                       const float* __restrict__ gt,     // (B,16,40)
                       const int*   __restrict__ lp,     // (B,16)
                       _Float16* __restrict__ xh, int t) {
    __shared__ float satt[HWN];
    __shared__ float sred[8];
    __shared__ float sb;
    const int tid = threadIdx.x, w = tid >> 5, lane = tid & 31;
    const int b = blockIdx.x;
    const float* hb = h + b * HID;

    // hdot = h . att_w_hid
    float v = hb[tid] * w_hid[tid] + hb[tid + 256] * w_hid[tid + 256];
    v = wave_sum(v);
    if (lane == 0) sred[w] = v;
    __syncthreads();
    if (tid == 0) { float s = 0; for (int i = 0; i < 8; ++i) s += sred[i]; sb = s; }
    __syncthreads();
    const float hdot = sb;
    __syncthreads();

    // softmax over n (tid == n), att = BETA * softmax
    float logit = simg[b * HWN + tid] + hdot;
    float m = wave_max(logit);
    if (lane == 0) sred[w] = m;
    __syncthreads();
    if (tid == 0) { float s = sred[0]; for (int i = 1; i < 8; ++i) s = fmaxf(s, sred[i]); sb = s; }
    __syncthreads();
    const float mx = sb;
    __syncthreads();
    float e = __expf(logit - mx);
    float es = wave_sum(e);
    if (lane == 0) sred[w] = es;
    __syncthreads();
    if (tid == 0) { float s = 0; for (int i = 0; i < 8; ++i) s += sred[i]; sb = s; }
    __syncthreads();
    satt[tid] = e * (2.0f / sb);   // BETA = 2.0
    __syncthreads();

    // pooled[c] = feat[b][c][:] . att ; wave w handles c in [w*64, w*64+64)
    const float* fb = feat + (size_t)b * CC * HWN;
    _Float16* xrow = xh + (size_t)b * KCMB;
    for (int c = w * 64; c < w * 64 + 64; ++c) {
        const float* fr = fb + c * HWN;
        float acc = 0.0f;
        #pragma unroll
        for (int i = 0; i < 8; ++i) acc += fr[lane + 32 * i] * satt[lane + 32 * i];
        acc = wave_sum(acc);
        if (lane == 0) xrow[c] = (_Float16)acc;
    }

    // scheduled-sampling input (cols 512..553)
    if (tid < OUTD) {
        float val;
        if (t == 0) {
            val = 0.0f;    // idx==0 forces use_prev, prev starts at zero
        } else {
            float rnd = jax_uniform_tb(t, b);
            if (rnd < 0.1f) {
                val = outs[((size_t)b * NL + (t - 1)) * OUTP + tid];
            } else if (tid < MAXPT) {
                val = gt[((size_t)b * NL + (t - 1)) * MAXPT + tid];
            } else {
                float pg = (float)lp[b * NL + (t - 1)];
                val = (tid == MAXPT) ? (1.0f - pg) : pg;
            }
        }
        xrow[CC + tid] = (_Float16)val;
    }
    // h copy (cols 554..1065) + zero pad (1066..1087)
    xrow[INL + tid]       = (_Float16)hb[tid];
    xrow[INL + 256 + tid] = (_Float16)hb[256 + tid];
    if (tid < KCMB - INL - HID) xrow[INL + HID + tid] = (_Float16)0.0f;
}

// ---------------------------------------------------------------------------
// WMMA GEMMs. 8 waves/block; each wave computes a 16 x (NT*16) output panel,
// reusing one A fragment per K-step across NT wmma ops.
// ---------------------------------------------------------------------------
__global__ void gemm_gates_k(const _Float16* __restrict__ xh,
                             const _Float16* __restrict__ wc,
                             float* __restrict__ gates) {
    const int gid = blockIdx.x * 8 + (threadIdx.x >> 5);    // 256 waves
    const int m0  = (gid >> 5) << 4;                        // 8 M-tiles
    const int n0  = (gid & 31) << 6;                        // 32 groups of 64 cols
    v8f acc[4];
    wmma_panel<4>(xh, wc, m0, n0, KCMB, KCMB, KCMB, acc);
    const int lane = threadIdx.x & 31;
    const int rofs = (lane < 16) ? 0 : 8;
    #pragma unroll
    for (int j = 0; j < 4; ++j) {
        const int col = n0 + j * 16 + (lane & 15);
        #pragma unroll
        for (int r = 0; r < 8; ++r)
            gates[(size_t)(m0 + r + rofs) * N4H + col] = acc[j][r];
    }
}

__global__ void gemm_lin1_k(const _Float16* __restrict__ hrelu,
                            const _Float16* __restrict__ l1h,
                            const float* __restrict__ l1b,
                            _Float16* __restrict__ zrelu) {
    const int gid = blockIdx.x * 8 + (threadIdx.x >> 5);    // 64 waves
    const int m0  = (gid >> 3) << 4;
    const int n0  = (gid & 7) << 6;
    v8f acc[4];
    wmma_panel<4>(hrelu, l1h, m0, n0, HID, HID, HID, acc);
    const int lane = threadIdx.x & 31;
    const int rofs = (lane < 16) ? 0 : 8;
    #pragma unroll
    for (int j = 0; j < 4; ++j) {
        const int col = n0 + j * 16 + (lane & 15);
        const float bias = l1b[col];
        #pragma unroll
        for (int r = 0; r < 8; ++r)
            zrelu[(size_t)(m0 + r + rofs) * HID + col] =
                (_Float16)fmaxf(acc[j][r] + bias, 0.0f);
    }
}

__global__ void gemm_lin2_k(const _Float16* __restrict__ zrelu,
                            const _Float16* __restrict__ l2h,
                            const float* __restrict__ l2b,
                            float* __restrict__ outs, int t) {
    const int gid = threadIdx.x >> 5;                       // 8 waves, 1 block
    const int m0  = gid << 4;
    v8f acc[3];
    wmma_panel<3>(zrelu, l2h, m0, 0, HID, HID, HID, acc);
    const int lane = threadIdx.x & 31;
    const int rofs = (lane < 16) ? 0 : 8;
    #pragma unroll
    for (int j = 0; j < 3; ++j) {
        const int col = j * 16 + (lane & 15);
        if (col < OUTD) {
            const float bias = l2b[col];
            #pragma unroll
            for (int r = 0; r < 8; ++r) {
                int b = m0 + r + rofs;
                outs[((size_t)b * NL + t) * OUTP + col] = acc[j][r] + bias;
            }
        }
    }
}

// ---------------------------------------------------------------------------
// LSTM elementwise: gates (B x 4H) -> c,h update; also relu(h) as f16
// ---------------------------------------------------------------------------
__device__ __forceinline__ float sigf(float x) { return 1.0f / (1.0f + __expf(-x)); }

__global__ void lstm_k(const float* __restrict__ gates,
                       const float* __restrict__ b_ih,
                       const float* __restrict__ b_hh,
                       float* __restrict__ c, float* __restrict__ h,
                       _Float16* __restrict__ hrelu) {
    int idx = blockIdx.x * 256 + threadIdx.x;       // B*HID
    if (idx >= BB * HID) return;
    int b = idx >> 9, j = idx & (HID - 1);
    const float* g = gates + (size_t)b * N4H;
    float gi = g[j]            + b_ih[j]            + b_hh[j];
    float gf = g[HID + j]      + b_ih[HID + j]      + b_hh[HID + j];
    float gg = g[2 * HID + j]  + b_ih[2 * HID + j]  + b_hh[2 * HID + j];
    float go = g[3 * HID + j]  + b_ih[3 * HID + j]  + b_hh[3 * HID + j];
    float cn = sigf(gf) * c[idx] + sigf(gi) * tanhf(gg);
    float hn = sigf(go) * tanhf(cn);
    c[idx] = cn;
    h[idx] = hn;
    hrelu[idx] = (_Float16)fmaxf(hn, 0.0f);
}

// ---------------------------------------------------------------------------
// Final packing: point_pos (B*16*40) then softmax pairs (B*16, 2)
// ---------------------------------------------------------------------------
__global__ void pack_k(const float* __restrict__ outs, float* __restrict__ out) {
    int idx = blockIdx.x * 256 + threadIdx.x;       // B*NL rows
    if (idx >= BB * NL) return;
    const float* row = outs + (size_t)idx * OUTP;
    float* pp = out + (size_t)idx * MAXPT;
    #pragma unroll
    for (int k = 0; k < MAXPT; ++k) pp[k] = row[k];
    float a = row[MAXPT], b = row[MAXPT + 1];
    float m = fmaxf(a, b);
    float ea = __expf(a - m), eb = __expf(b - m);
    float inv = 1.0f / (ea + eb);
    float* lpout = out + (size_t)BB * NL * MAXPT + (size_t)idx * 2;
    lpout[0] = ea * inv;
    lpout[1] = eb * inv;
}

// ---------------------------------------------------------------------------
// Host-side launch
// ---------------------------------------------------------------------------
extern "C" void kernel_launch(void* const* d_in, const int* in_sizes, int n_in,
                              void* d_out, int out_size, void* d_ws, size_t ws_size,
                              hipStream_t stream) {
    (void)in_sizes; (void)n_in; (void)out_size; (void)ws_size;
    const float* img   = (const float*)d_in[0];
    const float* gt    = (const float*)d_in[1];
    const int*   lp    = (const int*)d_in[2];
    const float* w_img = (const float*)d_in[4];
    const float* w_hid = (const float*)d_in[5];
    const float* att_b = (const float*)d_in[6];
    const float* w_ih  = (const float*)d_in[7];
    const float* w_hh  = (const float*)d_in[8];
    const float* b_ih  = (const float*)d_in[9];
    const float* b_hh  = (const float*)d_in[10];
    const float* l1w   = (const float*)d_in[11];
    const float* l1b   = (const float*)d_in[12];
    const float* l2w   = (const float*)d_in[13];
    const float* l2b   = (const float*)d_in[14];
    float* out = (float*)d_out;

    char* p = (char*)d_ws;
    auto alloc = [&](size_t bytes) {
        char* r = p;
        p += (bytes + 255) & ~(size_t)255;
        return r;
    };
    _Float16* wc    = (_Float16*)alloc((size_t)N4H * KCMB * 2);
    _Float16* l1h   = (_Float16*)alloc((size_t)HID * HID * 2);
    _Float16* l2h   = (_Float16*)alloc((size_t)OUTP * HID * 2);
    float*    simg  = (float*)   alloc((size_t)BB * HWN * 4);
    _Float16* xh    = (_Float16*)alloc((size_t)BB * KCMB * 2);
    float*    gates = (float*)   alloc((size_t)BB * N4H * 4);
    float*    h     = (float*)   alloc((size_t)BB * HID * 4);
    float*    c     = (float*)   alloc((size_t)BB * HID * 4);
    _Float16* hrelu = (_Float16*)alloc((size_t)BB * HID * 2);
    _Float16* zrelu = (_Float16*)alloc((size_t)BB * HID * 2);
    float*    outs  = (float*)   alloc((size_t)BB * NL * OUTP * 4);

    // one-time prep (re-run every call for determinism)
    conv_wc_k <<<(N4H * KCMB + 255) / 256, 256, 0, stream>>>(w_ih, w_hh, wc);
    conv_f16_k<<<(HID * HID + 255) / 256, 256, 0, stream>>>(l1w, l1h, HID * HID);
    conv_l2_k <<<(OUTP * HID + 255) / 256, 256, 0, stream>>>(l2w, l2h);
    simg_k    <<<BB, HWN, 0, stream>>>(img, w_img, att_b, simg);
    zero_k    <<<(BB * HID + 255) / 256, 256, 0, stream>>>(h, BB * HID);
    zero_k    <<<(BB * HID + 255) / 256, 256, 0, stream>>>(c, BB * HID);

    // sequential scan over 16 steps; kernel boundaries give the dependency
    for (int t = 0; t < NL; ++t) {
        attn_k      <<<BB, 256, 0, stream>>>(img, simg, w_hid, h, outs, gt, lp, xh, t);
        gemm_gates_k<<<32, 256, 0, stream>>>(xh, wc, gates);       // 256 waves x 4 tiles
        lstm_k      <<<(BB * HID + 255) / 256, 256, 0, stream>>>(gates, b_ih, b_hh, c, h, hrelu);
        gemm_lin1_k <<<8, 256, 0, stream>>>(hrelu, l1h, l1b, zrelu);
        gemm_lin2_k <<<1, 256, 0, stream>>>(zrelu, l2h, l2b, outs, t);
    }
    pack_k<<<(BB * NL + 255) / 256, 256, 0, stream>>>(outs, out);
}
